// GCN_21534966022931
// MI455X (gfx1250) — matrix operand order repaired and checked by reading, not compiled, not run on hardware
//
#include <hip/hip_runtime.h>

// Problem constants (match reference)
#define NUu 100000
#define NIi 50000
#define DD  64
#define LLa 2
#define EBe 500000
#define ECe 1000000
#define EPe 2000000

typedef __attribute__((ext_vector_type(16))) _Float16 v16h;
typedef __attribute__((ext_vector_type(8)))  float    v8f;

// ---------------- elementwise helpers (n must be a multiple of 4) ----------------

__global__ __launch_bounds__(256) void k_zero(float* __restrict__ p, long long n4) {
  long long i = ((long long)blockIdx.x * 256 + threadIdx.x) * 4;
  if (i < n4) *(float4*)(p + i) = make_float4(0.f, 0.f, 0.f, 0.f);
}

__global__ __launch_bounds__(256) void k_scale(float* __restrict__ out,
                                               const float* __restrict__ in,
                                               float s, long long n4) {
  long long i = ((long long)blockIdx.x * 256 + threadIdx.x) * 4;
  if (i < n4) {
    float4 v = *(const float4*)(in + i);
    v.x *= s; v.y *= s; v.z *= s; v.w *= s;
    *(float4*)(out + i) = v;
  }
}

__global__ __launch_bounds__(256) void k_axpy(float* __restrict__ a,
                                              const float* __restrict__ b,
                                              long long n4) {
  long long i = ((long long)blockIdx.x * 256 + threadIdx.x) * 4;
  if (i < n4) {
    float4 va = *(float4*)(a + i);
    float4 vb = *(const float4*)(b + i);
    va.x += vb.x; va.y += vb.y; va.z += vb.z; va.w += vb.w;
    *(float4*)(a + i) = va;
  }
}

__global__ __launch_bounds__(256) void k_clamp1(float* __restrict__ p, long long n) {
  long long i = (long long)blockIdx.x * 256 + threadIdx.x;
  if (i < n) p[i] = fmaxf(p[i], 1.0f);
}

// ---------------- graph kernels ----------------

__global__ __launch_bounds__(256)
void k_deg(const int* __restrict__ u, const int* __restrict__ v,
           float* __restrict__ du, float* __restrict__ dv, int E) {
  int e = blockIdx.x * 256 + threadIdx.x;
  if (e >= E) return;
  atomicAdd(&du[u[e]], 1.0f);
  atomicAdd(&dv[v[e]], 1.0f);
}

// msg = (feat[u] * deg(u)^-0.5) * edge_w  scatter-added into agg[v]
// one thread = one edge x 4 features
__global__ __launch_bounds__(256)
void k_scatter_msg(const int* __restrict__ u, const int* __restrict__ v,
                   const float* __restrict__ feat, const float* __restrict__ ew,
                   const float* __restrict__ degSrc, float* __restrict__ agg, int E) {
  long long t = (long long)blockIdx.x * 256 + threadIdx.x;
  int e = (int)(t >> 4);
  if (e >= E) return;
  int f = (int)(t & 15) * 4;
  int su = u[e], dv = v[e];
  float on = rsqrtf(degSrc[su]);
  float4 fs = *(const float4*)(feat + (long long)su * 64 + f);
  float4 w4 = *(const float4*)(ew + (long long)e * 64 + f);
  float* ap = agg + (long long)dv * 64 + f;
  atomicAdd(ap + 0, fs.x * on * w4.x);
  atomicAdd(ap + 1, fs.y * on * w4.y);
  atomicAdd(ap + 2, fs.z * on * w4.z);
  atomicAdd(ap + 3, fs.w * on * w4.w);
}

// se[u] += edge ; de[v] += edge
__global__ __launch_bounds__(256)
void k_scatter_esum(const int* __restrict__ u, const int* __restrict__ v,
                    const float* __restrict__ ew, float* __restrict__ se,
                    float* __restrict__ de, int E) {
  long long t = (long long)blockIdx.x * 256 + threadIdx.x;
  int e = (int)(t >> 4);
  if (e >= E) return;
  int f = (int)(t & 15) * 4;
  float4 w4 = *(const float4*)(ew + (long long)e * 64 + f);
  float* sp = se + (long long)u[e] * 64 + f;
  float* dp = de + (long long)v[e] * 64 + f;
  atomicAdd(sp + 0, w4.x); atomicAdd(sp + 1, w4.y);
  atomicAdd(sp + 2, w4.z); atomicAdd(sp + 3, w4.w);
  atomicAdd(dp + 0, w4.x); atomicAdd(dp + 1, w4.y);
  atomicAdd(dp + 2, w4.z); atomicAdd(dp + 3, w4.w);
}

// edge (in place) = (se[u] + de[v]) / (sd[u] + dd[v])
__global__ __launch_bounds__(256)
void k_temp(const int* __restrict__ u, const int* __restrict__ v,
            const float* __restrict__ degU, const float* __restrict__ degV,
            const float* __restrict__ se, const float* __restrict__ de,
            float* __restrict__ ew, int E) {
  long long t = (long long)blockIdx.x * 256 + threadIdx.x;
  int e = (int)(t >> 4);
  if (e >= E) return;
  int f = (int)(t & 15) * 4;
  int su = u[e], dv = v[e];
  float inv = 1.0f / (degU[su] + degV[dv]);
  float4 s4 = *(const float4*)(se + (long long)su * 64 + f);
  float4 d4 = *(const float4*)(de + (long long)dv * 64 + f);
  float4 o;
  o.x = (s4.x + d4.x) * inv; o.y = (s4.y + d4.y) * inv;
  o.z = (s4.z + d4.z) * inv; o.w = (s4.w + d4.w) * inv;
  *(float4*)(ew + (long long)e * 64 + f) = o;
}

// ---------------- fused WMMA GEMM: out = op(A[rows,64] @ W[64,64]) ----------------
// NORM:  val *= rsqrt(degDst[row]) before LeakyReLU (gconv in_norm)
// ACCUM: out += wscale * leaky(val)  else  out = leaky(val)
// rows is a multiple of 16. Block = 256 threads = 8 waves; wave owns 16 rows x 64 cols.
// W is staged in LDS pre-packed in B-fragment order: one contiguous v16h per
// (kstep, ntile, lane) so the hot loop issues ds_load_b128 pairs, not ds_load_u16.
template <bool NORM, bool ACCUM>
__global__ __launch_bounds__(256)
void k_gemm64(const float* __restrict__ A, const float* __restrict__ W,
              float* __restrict__ out, const float* __restrict__ degDst,
              float wscale, int nrows) {
  __shared__ _Float16 Ws[2 * 4 * 32 * 16];  // [kstep][ntile][lane][16] = 8 KB
  {
    // each of the 256 threads packs exactly one B fragment
    const int t  = threadIdx.x;
    const int ks = t >> 7, nt = (t >> 5) & 3, ln = t & 31;
    const int hf = ln >> 4, nn = ln & 15;
    v16h frag;
#pragma unroll
    for (int p = 0; p < 8; ++p) {
      const int k = ks * 32 + 8 * hf + 2 * p + (p >= 4 ? 8 : 0);
      frag[2 * p]     = (_Float16)W[k * 64 + nt * 16 + nn];
      frag[2 * p + 1] = (_Float16)W[(k + 1) * 64 + nt * 16 + nn];
    }
    *(v16h*)&Ws[(size_t)t * 16] = frag;
  }
  __syncthreads();

  const int tid = threadIdx.x;
  const int wave = tid >> 5, lane = tid & 31;
  const int rowBase = (blockIdx.x * 8 + wave) * 16;
  if (rowBase >= nrows) return;  // wave-uniform, after the barrier

  const int m = lane & 15, half = lane >> 4, n0 = lane & 15;
  const int row = rowBase + m;

  // A fragments: for this lane the needed K values form four contiguous 8-float
  // runs at K = 32*s + 16*g + 8*half -> 8x global_load_b128 + packed cvt.
  const float4* A4 = (const float4*)(A + (long long)row * 64);
  v16h a0, a1;
#pragma unroll
  for (int g = 0; g < 2; ++g) {          // a0: element group g -> halfs 8g..8g+7
    float4 lo = A4[g * 4 + 2 * half + 0];
    float4 hi = A4[g * 4 + 2 * half + 1];
    a0[8 * g + 0] = (_Float16)lo.x; a0[8 * g + 1] = (_Float16)lo.y;
    a0[8 * g + 2] = (_Float16)lo.z; a0[8 * g + 3] = (_Float16)lo.w;
    a0[8 * g + 4] = (_Float16)hi.x; a0[8 * g + 5] = (_Float16)hi.y;
    a0[8 * g + 6] = (_Float16)hi.z; a0[8 * g + 7] = (_Float16)hi.w;
  }
#pragma unroll
  for (int g = 0; g < 2; ++g) {          // a1: same, K offset +32
    float4 lo = A4[8 + g * 4 + 2 * half + 0];
    float4 hi = A4[8 + g * 4 + 2 * half + 1];
    a1[8 * g + 0] = (_Float16)lo.x; a1[8 * g + 1] = (_Float16)lo.y;
    a1[8 * g + 2] = (_Float16)lo.z; a1[8 * g + 3] = (_Float16)lo.w;
    a1[8 * g + 4] = (_Float16)hi.x; a1[8 * g + 5] = (_Float16)hi.y;
    a1[8 * g + 6] = (_Float16)hi.z; a1[8 * g + 7] = (_Float16)hi.w;
  }

  float norm[8];
#pragma unroll
  for (int r = 0; r < 8; ++r) {
    const int rr = rowBase + r + 8 * half;
    norm[r] = NORM ? rsqrtf(degDst[rr]) : 1.0f;
  }

#pragma unroll
  for (int nt = 0; nt < 4; ++nt) {
    v8f c = {};
    v16h b0 = *(const v16h*)&Ws[(size_t)((0 * 4 + nt) * 32 + lane) * 16];
    v16h b1 = *(const v16h*)&Ws[(size_t)((1 * 4 + nt) * 32 + lane) * 16];
    c = __builtin_amdgcn_wmma_f32_16x16x32_f16(false, a0, false, b0, (short)0, c, false, false);
    c = __builtin_amdgcn_wmma_f32_16x16x32_f16(false, a1, false, b1, (short)0, c, false, false);

#pragma unroll
    for (int r = 0; r < 8; ++r) {
      const int rr = rowBase + r + 8 * half;
      float v = NORM ? c[r] * norm[r] : c[r];
      v = (v >= 0.f) ? v : 0.01f * v;  // LeakyReLU
      float* dp = &out[(long long)rr * 64 + nt * 16 + n0];
      if (ACCUM) {
        *dp += wscale * v;
      } else {
        *dp = v;
      }
    }
  }
}

// ---------------- host orchestration ----------------

static inline unsigned grid4(long long n) { return (unsigned)((n / 4 + 255) / 256); }
static inline unsigned gridE16(long long E) { return (unsigned)((E * 16 + 255) / 256); }

extern "C" void kernel_launch(void* const* d_in, const int* in_sizes, int n_in,
                              void* d_out, int out_size, void* d_ws, size_t ws_size,
                              hipStream_t stream) {
  (void)in_sizes; (void)n_in; (void)out_size; (void)ws_size;

  const float* user_emb = (const float*)d_in[0];
  const float* item_emb = (const float*)d_in[1];
  const float* ebuy_in  = (const float*)d_in[2];
  const float* ecart_in = (const float*)d_in[3];
  const float* epv_in   = (const float*)d_in[4];
  const float* node_W   = (const float*)d_in[5];
  const float* edge_W   = (const float*)d_in[6];
  const int* buy_u  = (const int*)d_in[7];
  const int* buy_v  = (const int*)d_in[8];
  const int* cart_u = (const int*)d_in[9];
  const int* cart_v = (const int*)d_in[10];
  const int* pv_u   = (const int*)d_in[11];
  const int* pv_v   = (const int*)d_in[12];

  float* ws = (float*)d_ws;
  size_t o = 0;
  auto take = [&](size_t n) { float* p = ws + o; o += n; return p; };

  float* srcP[2] = { take((size_t)NUu * 64), take((size_t)NUu * 64) };
  float* dstP[2] = { take((size_t)NIi * 64), take((size_t)NIi * 64) };
  float* src_all = take((size_t)NUu * 64);
  float* dst_all = take((size_t)NIi * 64);
  float* eb = take((size_t)EBe * 64);
  float* ec = take((size_t)ECe * 64);
  float* ep = take((size_t)EPe * 64);
  float* agg = take((size_t)NUu * 64);           // big enough for NU or NI rows
  float* se  = take((size_t)NUu * 64);
  float* de  = take((size_t)NIi * 64);
  float* degs = take(3 * ((size_t)NUu + NIi));   // contiguous deg region
  float* dU[3] = { degs, degs + (NUu + NIi), degs + 2 * ((size_t)NUu + NIi) };
  float* dV[3] = { dU[0] + NUu, dU[1] + NUu, dU[2] + NUu };

  const float Asc = 0.0045f, Bsc = 0.0045f;

  // ---- degrees (graph-constant) ----
  long long ndeg = 3LL * (NUu + NIi);
  k_zero<<<grid4(ndeg), 256, 0, stream>>>(degs, ndeg);
  k_deg<<<(EBe + 255) / 256, 256, 0, stream>>>(buy_u,  buy_v,  dU[0], dV[0], EBe);
  k_deg<<<(ECe + 255) / 256, 256, 0, stream>>>(cart_u, cart_v, dU[1], dV[1], ECe);
  k_deg<<<(EPe + 255) / 256, 256, 0, stream>>>(pv_u,   pv_v,   dU[2], dV[2], EPe);
  k_clamp1<<<(unsigned)((ndeg + 255) / 256), 256, 0, stream>>>(degs, ndeg);

  // ---- scaled inputs ----
  k_scale<<<grid4((long long)NUu * 64), 256, 0, stream>>>(srcP[0], user_emb, Asc, (long long)NUu * 64);
  k_scale<<<grid4((long long)NIi * 64), 256, 0, stream>>>(dstP[0], item_emb, Asc, (long long)NIi * 64);
  k_scale<<<grid4((long long)NUu * 64), 256, 0, stream>>>(src_all, user_emb, Asc, (long long)NUu * 64);
  k_scale<<<grid4((long long)NIi * 64), 256, 0, stream>>>(dst_all, item_emb, Asc, (long long)NIi * 64);
  k_scale<<<grid4((long long)EBe * 64), 256, 0, stream>>>(eb, ebuy_in,  Bsc, (long long)EBe * 64);
  k_scale<<<grid4((long long)ECe * 64), 256, 0, stream>>>(ec, ecart_in, Bsc, (long long)ECe * 64);
  k_scale<<<grid4((long long)EPe * 64), 256, 0, stream>>>(ep, epv_in,   Bsc, (long long)EPe * 64);

  struct Rel { const int* u; const int* v; float* e; int E; float w; float* du; float* dv; };
  Rel rel[3] = {
    { buy_u,  buy_v,  eb, EBe, 1.00f, dU[0], dV[0] },
    { cart_u, cart_v, ec, ECe, 0.50f, dU[1], dV[1] },
    { pv_u,   pv_v,   ep, EPe, 0.25f, dU[2], dV[2] },
  };

  int cur = 0;
  for (int layer = 0; layer < LLa; ++layer) {
    const float* Wn = node_W + (size_t)layer * 64 * 64;
    const float* We = edge_W + (size_t)layer * 64 * 64;
    float* s_cur = srcP[cur], *d_cur = dstP[cur];
    float* s_new = srcP[1 - cur], *d_new = dstP[1 - cur];

    k_zero<<<grid4((long long)NUu * 64), 256, 0, stream>>>(s_new, (long long)NUu * 64);
    k_zero<<<grid4((long long)NIi * 64), 256, 0, stream>>>(d_new, (long long)NIi * 64);

    // node updates: all gconvs read the old src/dst and old edge features
    for (int r = 0; r < 3; ++r) {
      // forward: users -> items
      k_zero<<<grid4((long long)NIi * 64), 256, 0, stream>>>(agg, (long long)NIi * 64);
      k_scatter_msg<<<gridE16(rel[r].E), 256, 0, stream>>>(
          rel[r].u, rel[r].v, s_cur, rel[r].e, rel[r].du, agg, rel[r].E);
      k_gemm64<true, true><<<(NIi + 127) / 128, 256, 0, stream>>>(
          agg, Wn, d_new, rel[r].dv, rel[r].w, NIi);
      // reverse: items -> users
      k_zero<<<grid4((long long)NUu * 64), 256, 0, stream>>>(agg, (long long)NUu * 64);
      k_scatter_msg<<<gridE16(rel[r].E), 256, 0, stream>>>(
          rel[r].v, rel[r].u, d_cur, rel[r].e, rel[r].dv, agg, rel[r].E);
      k_gemm64<true, true><<<(NUu + 127) / 128, 256, 0, stream>>>(
          agg, Wn, s_new, rel[r].du, rel[r].w, NUu);
    }

    // edge updates (in place: new edge depends only on se/de/degrees)
    for (int r = 0; r < 3; ++r) {
      k_zero<<<grid4((long long)NUu * 64), 256, 0, stream>>>(se, (long long)NUu * 64);
      k_zero<<<grid4((long long)NIi * 64), 256, 0, stream>>>(de, (long long)NIi * 64);
      k_scatter_esum<<<gridE16(rel[r].E), 256, 0, stream>>>(
          rel[r].u, rel[r].v, rel[r].e, se, de, rel[r].E);
      k_temp<<<gridE16(rel[r].E), 256, 0, stream>>>(
          rel[r].u, rel[r].v, rel[r].du, rel[r].dv, se, de, rel[r].e, rel[r].E);
      k_gemm64<false, false><<<(unsigned)((rel[r].E + 127) / 128), 256, 0, stream>>>(
          rel[r].e, We, rel[r].e, nullptr, 1.0f, rel[r].E);
    }

    k_axpy<<<grid4((long long)NUu * 64), 256, 0, stream>>>(src_all, s_new, (long long)NUu * 64);
    k_axpy<<<grid4((long long)NIi * 64), 256, 0, stream>>>(dst_all, d_new, (long long)NIi * 64);
    cur = 1 - cur;
  }

  const float invL1 = 1.0f / (LLa + 1);
  float* outp = (float*)d_out;
  k_scale<<<grid4((long long)NUu * 64), 256, 0, stream>>>(outp, src_all, invL1, (long long)NUu * 64);
  k_scale<<<grid4((long long)NIi * 64), 256, 0, stream>>>(outp + (size_t)NUu * 64, dst_all, invL1, (long long)NIi * 64);
}
